// GaussianNetwork__90615220011505
// MI455X (gfx1250) — matrix operand order repaired and checked by reading, compiled, not run on hardware
//
#include <hip/hip_runtime.h>
#include <math.h>

#define NPOINTS_PER_BLOCK 128
#define HASHMASK ((1u << 19) - 1u)

typedef __attribute__((ext_vector_type(16))) _Float16 v16h;
typedef __attribute__((ext_vector_type(8)))  float    v8f;
typedef __attribute__((ext_vector_type(4)))  float    f32x4;
typedef __attribute__((ext_vector_type(2)))  float    f32x2;

union F16Frag { v16h v; f32x4 q[2]; };

struct ResArr { int r[16]; };

// Branch-free stable softplus(beta=100) on raw v_exp_f32/v_log_f32 (base-2):
//   softplus100(h) = max(h,0) + ln2/100 * log2(1 + exp2(-100*log2(e)*|h|))
// log argument is in [1,2] -> no denorm fixup needed; matches the reference's
// thresholded form to within f32 rounding (correction < 2^-35 for h > 0.2).
__device__ __forceinline__ float softplus100(float h) {
    const float e    = __builtin_amdgcn_exp2f(-144.26950408889634f * fabsf(h));
    const float corr = 0.0069314718055994531f * __builtin_amdgcn_logf(1.0f + e);
    return fmaxf(h, 0.0f) + corr;
}

// store two floats as two packed f16 at a 4B-aligned LDS address
__device__ __forceinline__ void st2h(_Float16* p, float a, float b) {
    union { _Float16 h[2]; unsigned u; } v;
    v.h[0] = (_Float16)a; v.h[1] = (_Float16)b;
    *(unsigned*)p = v.u;
}

__global__ __launch_bounds__(128, 4) void sdf_fused_kernel(
    const float* __restrict__ x,
    const float* __restrict__ tables,
    const float* __restrict__ W0, const float* __restrict__ b0,
    const float* __restrict__ W1, const float* __restrict__ b1,
    const float* __restrict__ W2, const float* __restrict__ b2,
    float* __restrict__ out,
    ResArr res)
{
    // ---- LDS (64 KB total) ----
    // Feature/K column remap (pad col3 so all pairs are 4B aligned):
    //   c0..2 = xyz, c3 = 0, c4..39 = embedding, c40..71 = hash, c72..95 = 0
    // sW0 columns are permuted identically during staging.
    __shared__ _Float16 sW0[64][96];
    __shared__ _Float16 sW1[64][64];
    __shared__ _Float16 sW2[32][64];     // rows 17..31 zero
    __shared__ _Float16 sA[4][32][96];   // per-wave features; rows reused for layer1 output
    __shared__ _Float16 sH[4][32][64];   // per-wave layer0 activations

    const int tid  = threadIdx.x;
    const int wave = tid >> 5;
    const int lane = tid & 31;
    const int half = lane >> 4;   // 0: lanes 0-15, 1: lanes 16-31
    const int l15  = lane & 15;

    // ---- stage weights f32 -> f16 (with the column remap for W0) ----
    for (int i = tid; i < 64 * 96; i += 128) {
        int r = i / 96, c = i - r * 96;
        float w = 0.0f;
        if (c < 3)                 w = W0[r * 71 + c];
        else if (c >= 4 && c < 72) w = W0[r * 71 + (c - 1)];
        sW0[r][c] = (_Float16)w;
    }
    for (int i = tid; i < 64 * 64; i += 128)
        sW1[i >> 6][i & 63] = (_Float16)W1[i];
    for (int i = tid; i < 32 * 64; i += 128) {
        int r = i >> 6, c = i & 63;
        sW2[r][c] = (_Float16)(r < 17 ? W2[r * 64 + c] : 0.0f);
    }

    // ---- per-lane encoding of one point -> sA row (f16, packed stores) ----
    const int pt = blockIdx.x * NPOINTS_PER_BLOCK + tid;
    const float px = x[pt * 3 + 0];
    const float py = x[pt * 3 + 1];
    const float pz = x[pt * 3 + 2];

    _Float16* row = &sA[wave][lane][0];
    st2h(row + 0, px, py);
    st2h(row + 2, pz, 0.0f);

    // positional embedding: per freq m (2^m): sin xyz then cos xyz, at col 4+6m.
    // raw v_sin/v_cos take revolutions; pre-scale by 1/2pi once (x*2^m exact).
    {
        const float INV2PI = 0.15915494309189535f;
        float ux = px * INV2PI, uy = py * INV2PI, uz = pz * INV2PI;
        #pragma unroll
        for (int m = 0; m < 6; ++m) {
            const int k = 4 + 6 * m;
            const float sx = __builtin_amdgcn_sinf(ux), sy = __builtin_amdgcn_sinf(uy);
            const float sz = __builtin_amdgcn_sinf(uz);
            const float cx = __builtin_amdgcn_cosf(ux), cy = __builtin_amdgcn_cosf(uy);
            const float cz = __builtin_amdgcn_cosf(uz);
            st2h(row + k + 0, sx, sy);
            st2h(row + k + 2, sz, cx);
            st2h(row + k + 4, cy, cz);
            ux *= 2.0f; uy *= 2.0f; uz *= 2.0f;
        }
    }

    // multiresolution hash-grid trilinear lookup (64MB of tables is L2-resident)
    for (int l = 0; l < 16; ++l) {
        const int R = res.r[l];
        const float* tbl = tables + ((size_t)l << 20);   // l * 2^19 * 2 floats
        const float s = (float)(R - 1);
        const float fx = px * s, fy = py * s, fz = pz * s;
        const float gx = floorf(fx), gy = floorf(fy), gz = floorf(fz);
        const int ix = (int)gx, iy = (int)gy, iz = (int)gz;
        const float wx = fx - gx, wy = fy - gy, wz = fz - gz;
        const bool dense = ((long long)R * R * R) <= (long long)(1 << 19);
        float a0 = 0.0f, a1 = 0.0f;
        #pragma unroll
        for (int n = 0; n < 8; ++n) {
            const int ox = (n >> 2) & 1, oy = (n >> 1) & 1, oz = n & 1;
            const int cxi = (ix + ox < R - 1) ? ix + ox : R - 1;
            const int cyi = (iy + oy < R - 1) ? iy + oy : R - 1;
            const int czi = (iz + oz < R - 1) ? iz + oz : R - 1;
            unsigned idx;
            if (dense) {
                idx = (unsigned)(cxi + R * (cyi + R * czi));
            } else {
                idx = (((unsigned)cxi) ^ ((unsigned)cyi * 2654435761u)
                                       ^ ((unsigned)czi * 805459861u)) & HASHMASK;
            }
            const float wgt = (ox ? wx : 1.0f - wx) * (oy ? wy : 1.0f - wy)
                            * (oz ? wz : 1.0f - wz);
            const f32x2 t = *(const f32x2*)(tbl + ((size_t)idx << 1));
            a0 += wgt * t.x;
            a1 += wgt * t.y;
        }
        st2h(row + 40 + 2 * l, a0, a1);
    }
    {   // zero cols 72..95 with three 16B stores
        const f32x4 z = {0.f, 0.f, 0.f, 0.f};
        *(f32x4*)(row + 72) = z;
        *(f32x4*)(row + 80) = z;
        *(f32x4*)(row + 88) = z;
    }

    __syncthreads();   // weights staged; own-wave sA writes are in-order wrt own-wave reads

    // ---- bias preload (C-matrix layout: bias depends only on n = nt*16 + l15) ----
    float bias0[4], bias1[4], bias2[2];
    #pragma unroll
    for (int nt = 0; nt < 4; ++nt) {
        bias0[nt] = b0[nt * 16 + l15];
        bias1[nt] = b1[nt * 16 + l15];
    }
    bias2[0] = b2[l15];
    bias2[1] = (l15 == 0) ? b2[16] : 0.0f;

    // ---- MLP via v_wmma_f32_16x16x32_f16; each wave owns 2 M-tiles (32 points) ----
    const _Float16* Abase = &sA[wave][0][0];
    _Float16*       Hbase = &sH[wave][0][0];

    // A fragment (16x32): lane<16 row l15 K {kt*32+0..7, +16..23}; lane>=16 K {+8..15, +24..31}
    auto ldA = [&](const _Float16* base, int pitch, int r, int kt) -> v16h {
        F16Frag u;
        const _Float16* p = base + r * pitch + kt * 32 + half * 8;
        u.q[0] = *(const f32x4*)(p);
        u.q[1] = *(const f32x4*)(p + 16);
        return u.v;
    };
    // B fragment (32x16): lane<16 col (nt*16+l15) K {kt*32+0..7, +16..23}; lane>=16 K {+8..15, +24..31}
    auto ldB = [&](const _Float16* base, int pitch, int nt, int kt) -> v16h {
        F16Frag u;
        const _Float16* p = base + (nt * 16 + l15) * pitch + kt * 32 + half * 8;
        u.q[0] = *(const f32x4*)(p);
        u.q[1] = *(const f32x4*)(p + 16);
        return u.v;
    };

    #pragma unroll
    for (int mt = 0; mt < 2; ++mt) {
        const int arow = mt * 16 + l15;

        // ---- layer 0: [16x96] x [96x64] ----
        #pragma unroll
        for (int nt = 0; nt < 4; ++nt) {
            v8f acc;
            #pragma unroll
            for (int j = 0; j < 8; ++j) acc[j] = bias0[nt];
            #pragma unroll
            for (int kt = 0; kt < 3; ++kt) {
                v16h a = ldA(Abase, 96, arow, kt);
                v16h b = ldB(&sW0[0][0], 96, nt, kt);
                acc = __builtin_amdgcn_wmma_f32_16x16x32_f16(
                          false, a, false, b, (short)0, acc, false, false);
            }
            #pragma unroll
            for (int j = 0; j < 8; ++j) {
                // D layout: element (m = half*8 + j, n = nt*16 + l15)
                Hbase[(mt * 16 + half * 8 + j) * 64 + nt * 16 + l15] =
                    (_Float16)softplus100(acc[j]);
            }
        }

        // ---- layer 1: [16x64] x [64x64]; output reuses sA rows (cols 0..63) ----
        #pragma unroll
        for (int nt = 0; nt < 4; ++nt) {
            v8f acc;
            #pragma unroll
            for (int j = 0; j < 8; ++j) acc[j] = bias1[nt];
            #pragma unroll
            for (int kt = 0; kt < 2; ++kt) {
                v16h a = ldA(Hbase, 64, arow, kt);
                v16h b = ldB(&sW1[0][0], 64, nt, kt);
                acc = __builtin_amdgcn_wmma_f32_16x16x32_f16(
                          false, a, false, b, (short)0, acc, false, false);
            }
            #pragma unroll
            for (int j = 0; j < 8; ++j) {
                sA[wave][mt * 16 + half * 8 + j][nt * 16 + l15] =
                    (_Float16)softplus100(acc[j]);
            }
        }

        // ---- layer 2: [16x64] x [64x32], only n < 17 stored ----
        #pragma unroll
        for (int nt = 0; nt < 2; ++nt) {
            const int n = nt * 16 + l15;
            v8f acc;
            #pragma unroll
            for (int j = 0; j < 8; ++j) acc[j] = bias2[nt];
            #pragma unroll
            for (int kt = 0; kt < 2; ++kt) {
                v16h a = ldA(Abase, 96, arow, kt);
                v16h b = ldB(&sW2[0][0], 64, nt, kt);
                acc = __builtin_amdgcn_wmma_f32_16x16x32_f16(
                          false, a, false, b, (short)0, acc, false, false);
            }
            if (n < 17) {
                const int pbase = blockIdx.x * NPOINTS_PER_BLOCK + wave * 32
                                + mt * 16 + half * 8;
                #pragma unroll
                for (int j = 0; j < 8; ++j)
                    __builtin_nontemporal_store(acc[j], &out[(size_t)(pbase + j) * 17 + n]);
            }
        }
    }
}

extern "C" void kernel_launch(void* const* d_in, const int* in_sizes, int n_in,
                              void* d_out, int out_size, void* d_ws, size_t ws_size,
                              hipStream_t stream) {
    const float* x  = (const float*)d_in[0];
    const float* tb = (const float*)d_in[1];
    const float* W0 = (const float*)d_in[2];
    const float* b0 = (const float*)d_in[3];
    const float* W1 = (const float*)d_in[4];
    const float* b1 = (const float*)d_in[5];
    const float* W2 = (const float*)d_in[6];
    const float* b2 = (const float*)d_in[7];
    float* out = (float*)d_out;

    // resolutions computed exactly like the numpy reference
    ResArr res;
    const double scale = exp2(log2(2048.0 / 16.0) / 15.0);
    for (int l = 0; l < 16; ++l)
        res.r[l] = (int)ceil(16.0 * pow(scale, (double)l));

    const int npts = in_sizes[0] / 3;          // 1048576
    const int blocks = npts / NPOINTS_PER_BLOCK;
    sdf_fused_kernel<<<blocks, NPOINTS_PER_BLOCK, 0, stream>>>(
        x, tb, W0, b0, W1, b1, W2, b2, out, res);
    (void)d_ws; (void)ws_size; (void)n_in; (void)out_size;
}
